// SimclrLoss_15229954032238
// MI455X (gfx1250) — matrix operand order repaired and checked by reading, not compiled, not run on hardware
//
#include <hip/hip_runtime.h>
#include <math.h>
#include <cstdint>

typedef __attribute__((ext_vector_type(16))) _Float16 v16h;
typedef __attribute__((ext_vector_type(8)))  _Float16 v8h;
typedef __attribute__((ext_vector_type(8)))  float    v8f;
typedef __attribute__((ext_vector_type(4)))  unsigned u32x4;
typedef __attribute__((ext_vector_type(4)))  int      i32x4;
typedef __attribute__((ext_vector_type(8)))  int      i32x8;

#define BS    4096
#define N2    8192        // 2*bs
#define D     256
#define INVT  2.0f        // 1/TEMPERATURE
#define MAXV  2.0f        // static bound: |logit| <= 2 for unit vectors
#define PANEL_ROWS  64
#define PANEL_HALFS (PANEL_ROWS * D)          // 16384 halfs (1<<14)
#define PANEL_BYTES (PANEL_HALFS * 2)         // 32 KB

#if __has_builtin(__builtin_amdgcn_tensor_load_to_lds) && \
    __has_builtin(__builtin_amdgcn_s_wait_tensorcnt)
#define USE_TDM 1
#else
#define USE_TDM 0
#endif

// ---------------------------------------------------------------------------
// Kernel 1: L2-normalize rows of zi/zj into fused f16 matrix Zh[N2][D].
// ---------------------------------------------------------------------------
__global__ void __launch_bounds__(256)
normalize_f16(const float* __restrict__ zi, const float* __restrict__ zj,
              _Float16* __restrict__ Zh) {
  const int row = blockIdx.x;
  const int t   = threadIdx.x;
  const float* src = (row < BS) ? (zi + (size_t)row * D)
                                : (zj + (size_t)(row - BS) * D);
  const float x = src[t];
  float ss = x * x;
  #pragma unroll
  for (int off = 16; off >= 1; off >>= 1) ss += __shfl_xor(ss, off, 32);
  __shared__ float warp_s[8];
  __shared__ float rnorm_s;
  if ((t & 31) == 0) warp_s[t >> 5] = ss;
  __syncthreads();
  if (t == 0) {
    float tot = 0.f;
    #pragma unroll
    for (int i = 0; i < 8; ++i) tot += warp_s[i];
    rnorm_s = rsqrtf(tot);
  }
  __syncthreads();
  Zh[(size_t)row * D + t] = (_Float16)(x * rnorm_s);
}

// ---------------------------------------------------------------------------
// WMMA fragment loaders (wave32 16x16x32 f16; contiguous k = 16*half + s, same for A and B).
// ---------------------------------------------------------------------------
__device__ __forceinline__ v16h load_a_frag(const _Float16* __restrict__ rowp,
                                            int q, int akoff) {
  const v8h lo = *(const v8h*)(rowp + q * 32 + akoff);
  const v8h hi = *(const v8h*)(rowp + q * 32 + akoff + 8);
  v16h r;
  #pragma unroll
  for (int i = 0; i < 8; ++i) { r[i] = lo[i]; r[i + 8] = hi[i]; }
  return r;
}

__device__ __forceinline__ v16h load_b_frag(const _Float16* __restrict__ colp,
                                            int q, int bkoff) {
  const v8h lo = *(const v8h*)(colp + q * 32 + bkoff);
  const v8h hi = *(const v8h*)(colp + q * 32 + bkoff + 8);
  v16h r;
  #pragma unroll
  for (int i = 0; i < 8; ++i) { r[i] = lo[i]; r[i + 8] = hi[i]; }
  return r;
}

// ---------------------------------------------------------------------------
// Stage a 64-row x 256-col f16 panel (rows c0..c0+63 of Zh) into LDS buffer.
// TDM path: wave 0 issues tensor_load_to_lds with a D# built per ISA ch.8.
// Fallback: cooperative global_load_b128 -> ds_store copy.
// ---------------------------------------------------------------------------
__device__ __forceinline__ void stage_panel(const _Float16* __restrict__ Zh,
                                            _Float16* __restrict__ lds_base_ptr,
                                            unsigned lds_byte_base,
                                            int c0, int buf, int tid) {
#if USE_TDM
  if ((tid >> 5) == 0) {   // one wave issues the DMA (wave-level instruction)
    const unsigned long long ga =
        (unsigned long long)(uintptr_t)(Zh + (size_t)c0 * D);
    u32x4 g0;
    g0[0] = 1u;                                              // count=1
    g0[1] = lds_byte_base + (unsigned)buf * PANEL_BYTES;     // lds_addr
    g0[2] = (unsigned)ga;                                    // global_addr lo
    g0[3] = ((unsigned)(ga >> 32) & 0x01FFFFFFu) | 0x80000000u; // hi | type=2
    i32x8 g1;
    g1[0] = 0x00010000;                 // data_size=1 (2 bytes/elem)
    g1[1] = (int)(D << 16);             // tensor_dim0[15:0] in bits 63:48
    g1[2] = (int)((unsigned)N2 << 16);  // tensor_dim0 hi | tensor_dim1 lo
    g1[3] = (int)(D << 16);             // tensor_dim1 hi | tile_dim0=256
    g1[4] = PANEL_ROWS;                 // tile_dim1=64 (tile_dim2=0)
    g1[5] = D;                          // tensor_dim0_stride lo
    g1[6] = 0;
    g1[7] = 0;
    i32x4 gz = {0, 0, 0, 0};
#if defined(__clang_major__) && __clang_major__ >= 23
    i32x8 gz8 = {0, 0, 0, 0, 0, 0, 0, 0};
    __builtin_amdgcn_tensor_load_to_lds(g0, g1, gz, gz, gz8, 0);
#else
    __builtin_amdgcn_tensor_load_to_lds(g0, g1, gz, gz, 0);
#endif
  }
  (void)lds_base_ptr;
#else
  (void)lds_byte_base;
  if (c0 + PANEL_ROWS < N2)
    __builtin_prefetch(Zh + (size_t)(c0 + PANEL_ROWS) * D, 0, 1);
  const v8h* src = (const v8h*)(Zh + (size_t)c0 * D);
  v8h*       dst = (v8h*)(lds_base_ptr + (unsigned)buf * PANEL_HALFS);
  #pragma unroll
  for (int i = 0; i < PANEL_HALFS / 8 / 256; ++i)
    dst[tid + i * 256] = src[tid + i * 256];
#endif
}

__device__ __forceinline__ void panel_sync(int tid) {
#if USE_TDM
  if ((tid >> 5) == 0) __builtin_amdgcn_s_wait_tensorcnt(0);
#endif
  __syncthreads();
}

// ---------------------------------------------------------------------------
// Kernel 2: fused S = Zh*Zh^T * invT with bounded-shift streaming sum-exp.
// Unit rows => every logit in [-2,2] => LSE = 2 + log(sum exp(v-2)):
// no running max, no rescale, 1 transcendental per element.
// Block = 8 waves, 32 rows; wave = 16x16 tile; 128 col panels of 64,
// double-buffered in LDS via TDM. All 8 B-fragments are preloaded into
// registers before the WMMA chain so LDS latency overlaps the matrix pipe.
// ---------------------------------------------------------------------------
__global__ void __launch_bounds__(256)
simclr_wmma_lse(const _Float16* __restrict__ Zh, float* __restrict__ row_out) {
  __shared__ __align__(16) _Float16 bbuf[2 * PANEL_HALFS];   // 64 KB
  __shared__ float sv[32][4], sp[32][4];

  const int tid  = threadIdx.x;
  const int lane = tid & 31;
  const int wave = tid >> 5;   // 0..7
  const int rt   = wave >> 2;  // row tile 0..1
  const int ct   = wave & 3;   // col tile 0..3
  const int r0   = blockIdx.x * 32;
  const int rowBase = r0 + rt * 16;

  const int lrow  = lane & 15;
  const int half  = lane >> 4;
  const int akoff = half << 4;   // 0 or 16 halfs
  const int bkoff = half << 4;   // 0 or 16 halfs

  // A fragments: this wave's 16 rows, full K=256, resident in registers.
  const _Float16* arow = Zh + (size_t)(rowBase + lrow) * D;
  v16h a[8];
  #pragma unroll
  for (int q = 0; q < 8; ++q) a[q] = load_a_frag(arow, q, akoff);

  const unsigned lds_byte_base = (unsigned)(uintptr_t)(void*)&bbuf[0];
  // Loop-invariant LDS tile offset (halfs); buffer select is pure arithmetic.
  const unsigned tileoff = (unsigned)(ct * 16 + lrow) * D;

  float s[8], pos[8];
  #pragma unroll
  for (int k = 0; k < 8; ++k) { s[k] = 0.f; pos[k] = 0.f; }

  // Prologue: stage panel 0.
  stage_panel(Zh, bbuf, lds_byte_base, 0, 0, tid);
  panel_sync(tid);

  for (int cb = 0; cb < 128; ++cb) {
    if (cb + 1 < 128)
      stage_panel(Zh, bbuf, lds_byte_base, (cb + 1) * PANEL_ROWS,
                  (cb + 1) & 1, tid);

    const _Float16* brow = bbuf + ((unsigned)(cb & 1) << 14) + tileoff;

    // Preload all 8 B fragments (16 ds_load_b128), then run the WMMA chain:
    // lets the scheduler retire LDS waits incrementally instead of dscnt==0
    // before every WMMA.
    v16h b[8];
    #pragma unroll
    for (int q = 0; q < 8; ++q) b[q] = load_b_frag(brow, q, bkoff);

    v8f acc = {};
    #pragma unroll
    for (int q = 0; q < 8; ++q)
      acc = __builtin_amdgcn_wmma_f32_16x16x32_f16(
          false, a[q], false, b[q], (short)0, acc, false, false);

    const int colTile = cb * 64 + ct * 16;   // 16-aligned col window
    // Diagonal / positive columns hit only when windows coincide (uniform).
    if (colTile != rowBase && colTile != (rowBase ^ BS)) {
      #pragma unroll
      for (int k = 0; k < 8; ++k)
        s[k] += __expf(fmaf(acc[k], INVT, -MAXV));
    } else {
      const int col_n = colTile + lrow;
      #pragma unroll
      for (int k = 0; k < 8; ++k) {
        const int row_m = rowBase + (half << 3) + k;
        const float t = fmaf(acc[k], INVT, -MAXV);   // v - 2
        if (col_n == (row_m ^ BS)) pos[k] += t;      // capture positive
        s[k] += (col_n == row_m) ? 0.f : __expf(t);  // exclude diagonal
      }
    }
    panel_sync(tid);
  }

  // Sum partials across the 16 lanes of each half-wave.
  #pragma unroll
  for (int k = 0; k < 8; ++k) {
    #pragma unroll
    for (int off = 8; off >= 1; off >>= 1) {
      s[k]   += __shfl_xor(s[k],   off, 16);
      pos[k] += __shfl_xor(pos[k], off, 16);
    }
  }

  // Cross-wave merge (4 column-waves per row) via LDS.
  if (lrow == 0) {
    #pragma unroll
    for (int k = 0; k < 8; ++k) {
      const int rloc = rt * 16 + (half << 3) + k;
      sv[rloc][ct] = s[k];
      sp[rloc][ct] = pos[k];
    }
  }
  __syncthreads();
  if (tid < 32) {
    float S = 0.f, P = 0.f;
    #pragma unroll
    for (int c = 0; c < 4; ++c) { S += sv[tid][c]; P += sp[tid][c]; }
    // LSE - v_pos = (MAXV + log S) - (P + MAXV) = log S - P
    row_out[r0 + tid] = __logf(S) - P;
  }
}

// ---------------------------------------------------------------------------
// Kernel 3: deterministic mean over row_out[N2] -> d_out[0].
// ---------------------------------------------------------------------------
__global__ void __launch_bounds__(256)
reduce_mean(const float* __restrict__ row_out, float* __restrict__ out) {
  const int t = threadIdx.x;
  float acc = 0.f;
  for (int i = t; i < N2; i += 256) acc += row_out[i];
  #pragma unroll
  for (int off = 16; off >= 1; off >>= 1) acc += __shfl_xor(acc, off, 32);
  __shared__ float w[8];
  if ((t & 31) == 0) w[t >> 5] = acc;
  __syncthreads();
  if (t == 0) {
    float stot = 0.f;
    #pragma unroll
    for (int i = 0; i < 8; ++i) stot += w[i];
    out[0] = stot / (float)N2;
  }
}

// ---------------------------------------------------------------------------
extern "C" void kernel_launch(void* const* d_in, const int* in_sizes, int n_in,
                              void* d_out, int out_size, void* d_ws, size_t ws_size,
                              hipStream_t stream) {
  const float* zi = (const float*)d_in[0];
  const float* zj = (const float*)d_in[1];
  _Float16* Zh     = (_Float16*)d_ws;                                  // 4 MB
  float*    rowout = (float*)((char*)d_ws + (size_t)N2 * D * sizeof(_Float16));
  float*    out    = (float*)d_out;

  normalize_f16  <<<N2,      256, 0, stream>>>(zi, zj, Zh);
  simclr_wmma_lse<<<N2 / 32, 256, 0, stream>>>(Zh, rowout);
  reduce_mean    <<<1,       256, 0, stream>>>(rowout, out);
}
